// SliceAttention_57999238365766
// MI455X (gfx1250) — matrix operand order repaired and checked by reading, not compiled, not run on hardware
//
#include <hip/hip_runtime.h>

// ---------------------------------------------------------------------------
// SliceAttention for MI455X (gfx1250, wave32, WMMA 16x16x32 bf16,
// async global->LDS staging with ASYNCcnt double buffering)
// ---------------------------------------------------------------------------

#define B_   8
#define N_   8192
#define D_   512
#define H_   8
#define M_   64
#define HD_  64
#define BN_  (B_ * N_)      // 65536 rows

typedef unsigned short ushort_t;
typedef __attribute__((ext_vector_type(16))) __bf16        v16bf;
typedef __attribute__((ext_vector_type(8)))  float         v8f;
typedef __attribute__((ext_vector_type(4)))  unsigned int  v4u;
typedef __attribute__((ext_vector_type(4)))  int           v4i;

union Frag {
    v16bf    v;
    v4u      q[2];
    ushort_t h[16];
};

__device__ __forceinline__ ushort_t f2bf(float f) {
    unsigned u = __builtin_bit_cast(unsigned, f);
    unsigned r = u + 0x7FFFu + ((u >> 16) & 1u);   // round-to-nearest-even
    return (ushort_t)(r >> 16);
}
__device__ __forceinline__ float bf2f(ushort_t h) {
    unsigned u = ((unsigned)h) << 16;
    return __builtin_bit_cast(float, u);
}

// ---------------------------------------------------------------------------
// CDNA5 async global->LDS copy (GLOBAL_LOAD_ASYNC_TO_LDS_B128, ASYNCcnt).
// Builtin signature (probe-confirmed by the round-2 diagnostic):
//   void __builtin_amdgcn_global_load_async_to_lds_b128(
//       v4i addrspace(1)* src, v4i addrspace(3)* dst, imm offset, imm cpol)
// Generic->AS(3) handoff via integer cast (low 32 bits of a generic LDS
// address are the LDS byte offset, ISA 10.2).  Guarded with __has_builtin so
// the file still compiles (plain LDS copy) on toolchains without it.
// ---------------------------------------------------------------------------
#if defined(__HIP_DEVICE_COMPILE__) && defined(__gfx1250__)
#if __has_builtin(__builtin_amdgcn_global_load_async_to_lds_b128)
#define HAVE_ASYNC_LDS 1
#endif
#endif

typedef __attribute__((address_space(1))) v4i* as1_v4i_p;
typedef __attribute__((address_space(3))) v4i* as3_v4i_p;

__device__ __forceinline__ void async_copy_b128(const void* g, void* l) {
#ifdef HAVE_ASYNC_LDS
    __builtin_amdgcn_global_load_async_to_lds_b128(
        (as1_v4i_p)(unsigned long long)g,
        (as3_v4i_p)(unsigned)(unsigned long long)l,
        0, 0);
#else
    *(v4u*)l = *(const v4u*)g;
#endif
}

template <int N>
__device__ __forceinline__ void async_lds_wait() {
#ifdef HAVE_ASYNC_LDS
#if __has_builtin(__builtin_amdgcn_s_wait_asynccnt)
    __builtin_amdgcn_s_wait_asynccnt(N);
#else
    asm volatile("s_wait_asynccnt %0" :: "i"(N) : "memory");
#endif
#endif
}

// ---------------------------------------------------------------------------
// Pack kernels: f32 -> bf16, straight or transposed
// ---------------------------------------------------------------------------
__global__ void pack_n_kernel(const float* __restrict__ src,
                              ushort_t* __restrict__ dst, long n) {
    long i = (long)blockIdx.x * 256 + threadIdx.x;
    if (i < n) dst[i] = f2bf(src[i]);
}

// dst[c][r] = bf16(src[r][c])   (src is rows x cols, row-major)
__global__ void pack_t_kernel(const float* __restrict__ src,
                              ushort_t* __restrict__ dst, int rows, int cols) {
    long i = (long)blockIdx.x * 256 + threadIdx.x;
    long n = (long)rows * cols;
    if (i < n) {
        int r = (int)(i / cols), c = (int)(i % cols);
        dst[(long)c * rows + r] = f2bf(src[(long)r * cols + c]);
    }
}

// ---------------------------------------------------------------------------
// Generic wave32 WMMA GEMM:  C[m][n] = sum_k A[m][k] * Bt[n][k]  (+ bias[n])
// A, Bt bf16 row-major.  The streamed A tile (BM x 32) is staged into LDS
// once per workgroup with double-buffered async copies (next K-tile copy
// overlaps current WMMAs); B fragments load direct from global (all B
// operands are small packed weights that live in the 192 MB L2).
// 8 waves/wg arranged WROWS x WCOLS, each wave owns WM x WN 16x16 tiles.
// Batched over blockIdx.z with (b,h) decomposition for head-strided operands.
// ---------------------------------------------------------------------------
template <int WM, int WN, int WROWS, int WCOLS, typename OutT, bool HAS_BIAS>
__global__ __launch_bounds__(256) void gemm_bt_kernel(
    const ushort_t* __restrict__ A,  long lda,
    const ushort_t* __restrict__ Bt, long ldb,
    OutT* __restrict__ C,            long ldc,
    const float* __restrict__ bias,
    int K, int batchH,
    long sAb, long sAh, long sBb, long sBh, long sCb, long sCh)
{
    static_assert(WROWS * WCOLS == 8, "8 waves per workgroup");
    constexpr int BM = WROWS * WM * 16;          // 64 or 128 rows per wg
    constexpr int CH = (BM * 32) / (256 * 8);    // 16B copies per thread/stage
    __shared__ __align__(16) ushort_t As[2][BM][32];

    int z = blockIdx.z;
    int b = z / batchH, h = z % batchH;
    A  += (long)b * sAb + (long)h * sAh;
    Bt += (long)b * sBb + (long)h * sBh;
    C  += (long)b * sCb + (long)h * sCh;

    const int  tid  = threadIdx.x;
    const int  lane = tid & 31;
    const int  wave = tid >> 5;
    const int  l15  = lane & 15;
    const int  hi   = lane >> 4;                 // 0 or 1
    const int  wr   = wave / WCOLS, wc = wave % WCOLS;
    const long mwg  = (long)blockIdx.x * BM;     // wg row base (global)
    const int  ml   = wr * (WM * 16);            // wave row base (in LDS tile)
    const long n0   = (long)blockIdx.y * (WCOLS * WN * 16) + (long)wc * (WN * 16);

    v8f acc[WM][WN];
#pragma unroll
    for (int i = 0; i < WM; i++)
#pragma unroll
        for (int j = 0; j < WN; j++)
#pragma unroll
            for (int e = 0; e < 8; e++) acc[i][j][e] = 0.0f;

    // stage A tile [BM][32] for K-step k0 into buffer buf
    auto stage = [&](int buf, int k0) {
#pragma unroll
        for (int c = 0; c < CH; c++) {
            int idx = c * 256 + tid;             // 16B chunk id
            int row = idx >> 2, seg = idx & 3;   // 4 x 16B per 64B row
            async_copy_b128(A + (mwg + row) * lda + k0 + seg * 8,
                            &As[buf][row][seg * 8]);
        }
    };

    const int nsteps = K >> 5;
    stage(0, 0);
    for (int i = 0; i < nsteps; i++) {
        int k0 = i << 5;
        bool has_next = (i + 1 < nsteps);
        if (has_next) stage((i + 1) & 1, k0 + 32);   // overlap copy w/ compute
        if (has_next) async_lds_wait<CH>();          // current tile landed
        else          async_lds_wait<0>();
        __syncthreads();

        Frag fa[WM], fb[WN];
#pragma unroll
        for (int ii = 0; ii < WM; ii++) {
            // A 16x32 bf16 frag: lane row, 16 contiguous k at hi*16
            const ushort_t* ap = &As[i & 1][ml + ii * 16 + l15][hi * 16];
            fa[ii].q[0] = *(const v4u*)ap;
            fa[ii].q[1] = *(const v4u*)(ap + 8);
        }
#pragma unroll
        for (int j = 0; j < WN; j++) {
            // B 32x16 bf16 frag: lane col, 16 contiguous k at k0+hi*16
            const ushort_t* bp = Bt + (n0 + j * 16 + l15) * ldb + k0 + hi * 16;
            fb[j].q[0] = *(const v4u*)bp;
            fb[j].q[1] = *(const v4u*)(bp + 8);
        }
#pragma unroll
        for (int ii = 0; ii < WM; ii++)
#pragma unroll
            for (int j = 0; j < WN; j++)
                acc[ii][j] = __builtin_amdgcn_wmma_f32_16x16x32_bf16(
                    false, fa[ii].v, false, fb[j].v, (short)0, acc[ii][j],
                    false, false);
        __syncthreads();
    }

#pragma unroll
    for (int i = 0; i < WM; i++)
#pragma unroll
        for (int j = 0; j < WN; j++)
#pragma unroll
            for (int r = 0; r < 8; r++) {
                long m = mwg + ml + i * 16 + r + hi * 8;  // C/D: VGPR r -> row
                long n = n0 + j * 16 + l15;
                float v = acc[i][j][r];
                if constexpr (HAS_BIAS) v += bias[n];
                if constexpr (__is_same(OutT, float)) C[m * ldc + n] = v;
                else                                  C[m * ldc + n] = f2bf(v);
            }
}

// ---------------------------------------------------------------------------
// Softmax over the M (slice) axis.  mixT is [BN][512] f32 with channel
// c = h*64 + m, so each head's 64 logits are contiguous. One wave per row,
// 16 channels/lane -> each head spans 4 lanes (xor-1/2 reductions).
// Writes bf16 slice weights.
// ---------------------------------------------------------------------------
__global__ __launch_bounds__(256) void softmax_m_kernel(
    const float* __restrict__ mixT, ushort_t* __restrict__ swT)
{
    int wave = threadIdx.x >> 5, lane = threadIdx.x & 31;
    long row = (long)blockIdx.x * 8 + wave;
    const float* src = mixT + row * 512 + lane * 16;
    float v[16];
#pragma unroll
    for (int i = 0; i < 16; i++) v[i] = src[i];
    float mx = -1e30f;
#pragma unroll
    for (int i = 0; i < 16; i++) mx = fmaxf(mx, v[i]);
    mx = fmaxf(mx, __shfl_xor(mx, 1, 32));
    mx = fmaxf(mx, __shfl_xor(mx, 2, 32));
    float s = 0.0f;
#pragma unroll
    for (int i = 0; i < 16; i++) { v[i] = __expf(v[i] - mx); s += v[i]; }
    s += __shfl_xor(s, 1, 32);
    s += __shfl_xor(s, 2, 32);
    float inv = 1.0f / s;
    union { v4u q[2]; ushort_t h[16]; } o;
#pragma unroll
    for (int i = 0; i < 16; i++) o.h[i] = f2bf(v[i] * inv);
    v4u* dst = (v4u*)(swT + row * 512 + lane * 16);
    dst[0] = o.q[0];
    dst[1] = o.q[1];
}

// ---------------------------------------------------------------------------
// slice_tok:  stok[bh][m][d] = sum_n slice_w[m][n] * xv[n][d], K = N = 8192.
// K-split over 8 chunks of 1024; A (slice weights, transposed layout) and
// B (values) are re-tiled through LDS; the slice-weight row sums (softmax
// denominator) are folded out of the already-loaded A fragments.
// Deterministic: per-chunk partials, reduced by a follow-up kernel.
// ---------------------------------------------------------------------------
__global__ __launch_bounds__(256) void slice_tok_kernel(
    const ushort_t* __restrict__ swT,   // [B][N][512] bf16, c = h*64+m
    const ushort_t* __restrict__ kvb,   // [B][N][1024] bf16, xv at 512+h*64+d
    float* __restrict__ stok_part,      // [8][64][64][64]
    float* __restrict__ sums_part)      // [8][64][64]
{
    const int CH = N_ / 8;              // 1024 n per chunk
    int chunk = blockIdx.x;
    int bh = blockIdx.z;
    int b = bh >> 3, h = bh & 7;

    __shared__ __align__(16) ushort_t As[64][32];   // [m][n]
    __shared__ __align__(16) ushort_t Bs[64][32];   // [d][n]

    int t = threadIdx.x;
    int lane = t & 31, wave = t >> 5, l15 = lane & 15, hi = lane >> 4;
    int tm = wave & 3, tng = wave >> 2;             // m-tile, n(d)-tile group

    v8f acc[2];
#pragma unroll
    for (int j = 0; j < 2; j++)
#pragma unroll
        for (int e = 0; e < 8; e++) acc[j][e] = 0.0f;
    float rs = 0.0f;

    int nl = t >> 3, mb = t & 7;                    // staging coords
    const ushort_t* aSrc = swT + (long)b * N_ * 512  + h * 64 + mb * 8;
    const ushort_t* bSrc = kvb + (long)b * N_ * 1024 + 512 + h * 64 + mb * 8;

    for (int kk = 0; kk < CH; kk += 32) {
        long nrow = (long)chunk * CH + kk + nl;
        union { v4u q; ushort_t s[8]; } ua, ub;
        ua.q = *(const v4u*)(aSrc + nrow * 512);
        ub.q = *(const v4u*)(bSrc + nrow * 1024);
#pragma unroll
        for (int q = 0; q < 8; q++) {
            As[mb * 8 + q][nl] = ua.s[q];           // transpose into [m][n]
            Bs[mb * 8 + q][nl] = ub.s[q];           // transpose into [d][n]
        }
        __syncthreads();

        Frag fa, fb0, fb1;
        const ushort_t* ar = &As[tm * 16 + l15][hi * 16];
        fa.q[0] = *(const v4u*)ar;
        fa.q[1] = *(const v4u*)(ar + 8);
        const ushort_t* b0 = &Bs[(tng * 2 + 0) * 16 + l15][hi * 16];
        fb0.q[0] = *(const v4u*)b0;
        fb0.q[1] = *(const v4u*)(b0 + 8);
        const ushort_t* b1 = &Bs[(tng * 2 + 1) * 16 + l15][hi * 16];
        fb1.q[0] = *(const v4u*)b1;
        fb1.q[1] = *(const v4u*)(b1 + 8);

        acc[0] = __builtin_amdgcn_wmma_f32_16x16x32_bf16(
            false, fa.v, false, fb0.v, (short)0, acc[0], false, false);
        acc[1] = __builtin_amdgcn_wmma_f32_16x16x32_bf16(
            false, fa.v, false, fb1.v, (short)0, acc[1], false, false);

        if (tng == 0) {                             // denominator: sum_n w[m][n]
#pragma unroll
            for (int q = 0; q < 16; q++) rs += bf2f(fa.h[q]);
        }
        __syncthreads();
    }

    long base = ((long)chunk * 64 + bh) * 4096;     // [chunk][bh][m][d]
#pragma unroll
    for (int jj = 0; jj < 2; jj++)
#pragma unroll
        for (int r = 0; r < 8; r++) {
            int m = tm * 16 + r + hi * 8;
            int d = (tng * 2 + jj) * 16 + l15;
            stok_part[base + (long)m * 64 + d] = acc[jj][r];
        }
    float tot = rs + __shfl_xor(rs, 16, 32);
    if (tng == 0 && hi == 0)
        sums_part[((long)chunk * 64 + bh) * 64 + tm * 16 + l15] = tot;
}

__global__ void reduce_parts_kernel(const float* __restrict__ stok_part,
                                    const float* __restrict__ sums_part,
                                    float* __restrict__ stok,
                                    float* __restrict__ sums)
{
    long i = (long)blockIdx.x * 256 + threadIdx.x;
    if (i < 262144) {
        float s = 0.0f;
#pragma unroll
        for (int c = 0; c < 8; c++) s += stok_part[(long)c * 262144 + i];
        stok[i] = s;
    }
    if (i < 4096) {
        float s = 0.0f;
#pragma unroll
        for (int c = 0; c < 8; c++) s += sums_part[(long)c * 4096 + i];
        sums[i] = s;
    }
}

// ---------------------------------------------------------------------------
// normalize by slice-weight sum + layernorm over d (HD=64); one wave per
// (b,h,m) row, 2 elements per lane.  Output bf16 in [b][m][h*64+d] layout.
// ---------------------------------------------------------------------------
__global__ __launch_bounds__(256) void norm_ln_kernel(
    const float* __restrict__ stok, const float* __restrict__ sums,
    const float* __restrict__ g, const float* __restrict__ be,
    ushort_t* __restrict__ stok_ln)
{
    int wave = threadIdx.x >> 5, lane = threadIdx.x & 31;
    long row = (long)blockIdx.x * 8 + wave;         // = bh*64 + m
    int bh = (int)(row >> 6), m = (int)(row & 63);
    int b = bh >> 3, h = bh & 7;
    int d0 = lane * 2;
    float inv = 1.0f / (sums[row] + 1e-5f);
    float t0 = stok[row * 64 + d0] * inv;
    float t1 = stok[row * 64 + d0 + 1] * inv;
    float mu = t0 + t1;
#pragma unroll
    for (int off = 1; off < 32; off <<= 1) mu += __shfl_xor(mu, off, 32);
    mu *= (1.0f / 64.0f);
    float a0 = t0 - mu, a1 = t1 - mu;
    float vr = a0 * a0 + a1 * a1;
#pragma unroll
    for (int off = 1; off < 32; off <<= 1) vr += __shfl_xor(vr, off, 32);
    vr *= (1.0f / 64.0f);
    float rn = rsqrtf(vr + 1e-5f);
    long dst = ((long)(b * 64 + m)) * 512 + h * 64 + d0;
    stok_ln[dst]     = f2bf(a0 * rn * g[d0]     + be[d0]);
    stok_ln[dst + 1] = f2bf(a1 * rn * g[d0 + 1] + be[d0 + 1]);
}

// ---------------------------------------------------------------------------
// mini self-attention over the 64 slice tokens, one workgroup per (b,h).
// Tiny (0.5 MFLOP/wg) -> plain VALU.  Emits out_tok TRANSPOSED ([d][m]) in
// bf16 so the scatter GEMM can load B fragments contiguously.
// ---------------------------------------------------------------------------
__global__ __launch_bounds__(256) void mini_attn_kernel(
    const float* __restrict__ qkvf,     // [B*M][1536]
    ushort_t* __restrict__ outT)        // [bh][d][m]
{
    int bh = blockIdx.x;
    int b = bh >> 3, h = bh & 7;
    __shared__ float qs[64][64];
    __shared__ float ks[64][64];
    __shared__ float vs[64][64];
    int t = threadIdx.x;
#pragma unroll
    for (int i = 0; i < 16; i++) {
        int e = t * 16 + i;
        int m = e >> 6, d = e & 63;
        long base = ((long)(b * 64 + m)) * 1536 + h * 64 + d;
        qs[m][d] = qkvf[base];
        ks[m][d] = qkvf[base + 512];
        vs[m][d] = qkvf[base + 1024];
    }
    __syncthreads();

    int r = t >> 2, qt = t & 3;                     // row, column quarter
    float dot[16];
#pragma unroll
    for (int j = 0; j < 16; j++) {
        int c = qt * 16 + j;
        float s = 0.0f;
        for (int d = 0; d < 64; d++) s += qs[r][d] * ks[c][d];
        dot[j] = s * 0.125f;                        // HD^-0.5
    }
    float mx = -1e30f;
#pragma unroll
    for (int j = 0; j < 16; j++) mx = fmaxf(mx, dot[j]);
    mx = fmaxf(mx, __shfl_xor(mx, 1, 32));
    mx = fmaxf(mx, __shfl_xor(mx, 2, 32));
    float sum = 0.0f;
#pragma unroll
    for (int j = 0; j < 16; j++) { dot[j] = __expf(dot[j] - mx); sum += dot[j]; }
    sum += __shfl_xor(sum, 1, 32);
    sum += __shfl_xor(sum, 2, 32);
    float inv = 1.0f / sum;
    __syncthreads();
#pragma unroll
    for (int j = 0; j < 16; j++) qs[r][qt * 16 + j] = dot[j] * inv;  // attn (reuse qs)
    __syncthreads();

    float acc[16];
#pragma unroll
    for (int j = 0; j < 16; j++) acc[j] = 0.0f;
    for (int c = 0; c < 64; c++) {
        float a = qs[r][c];
#pragma unroll
        for (int j = 0; j < 16; j++) acc[j] += a * vs[c][qt * 16 + j];
    }
#pragma unroll
    for (int j = 0; j < 16; j++) {
        int d = qt * 16 + j;
        outT[(long)bh * 4096 + (long)d * 64 + r] = f2bf(acc[j]);
    }
}

// ---------------------------------------------------------------------------
// Host-side orchestration
// ---------------------------------------------------------------------------
extern "C" void kernel_launch(void* const* d_in, const int* in_sizes, int n_in,
                              void* d_out, int out_size, void* d_ws, size_t ws_size,
                              hipStream_t stream)
{
    (void)in_sizes; (void)n_in; (void)out_size; (void)ws_size;
    const float* x     = (const float*)d_in[0];
    const float* W_kv  = (const float*)d_in[1];
    const float* b_kv  = (const float*)d_in[2];
    const float* wtq   = (const float*)d_in[3];
    const float* W_mix = (const float*)d_in[4];
    const float* ln_g  = (const float*)d_in[5];
    const float* ln_b  = (const float*)d_in[6];
    const float* W_qkv = (const float*)d_in[7];
    const float* W_out = (const float*)d_in[8];
    const float* b_out = (const float*)d_in[9];
    float* out = (float*)d_out;

    char* ws = (char*)d_ws;
    size_t off = 0;
    auto alloc = [&](size_t bytes) -> char* {
        char* p = ws + off;
        off = (off + bytes + 255) & ~(size_t)255;
        return p;
    };

    ushort_t* xb      = (ushort_t*)alloc((size_t)BN_ * D_ * 2);       //  64 MB
    ushort_t* kvb     = (ushort_t*)alloc((size_t)BN_ * 2 * D_ * 2);   // 128 MB
    ushort_t* scoresT = (ushort_t*)alloc((size_t)BN_ * 512 * 2);      //  64 MB
    float*    mixT    = (float*)   alloc((size_t)BN_ * 512 * 4);      // 128 MB
    ushort_t* Wkv_t   = (ushort_t*)alloc((size_t)1024 * 512 * 2);
    ushort_t* wtq_b   = (ushort_t*)alloc((size_t)H_ * M_ * HD_ * 2);
    ushort_t* Wmix_b  = (ushort_t*)alloc((size_t)512 * 512 * 2);
    ushort_t* Wqkv_t  = (ushort_t*)alloc((size_t)1536 * 512 * 2);
    ushort_t* Wout_t  = (ushort_t*)alloc((size_t)512 * 512 * 2);
    float*    stok_p  = (float*)   alloc((size_t)8 * 64 * 4096 * 4);  //   8 MB
    float*    sums_p  = (float*)   alloc((size_t)8 * 64 * 64 * 4);
    float*    stok    = (float*)   alloc((size_t)64 * 4096 * 4);
    float*    sums    = (float*)   alloc((size_t)64 * 64 * 4);
    ushort_t* stok_ln = (ushort_t*)alloc((size_t)B_ * M_ * D_ * 2);
    float*    qkvf    = (float*)   alloc((size_t)B_ * M_ * 1536 * 4);
    ushort_t* out_tT  = (ushort_t*)alloc((size_t)64 * 4096 * 2);
    // Buffer reuse (sequential pipeline): slice weights overwrite dead x_bf16,
    // the gathered output overwrites dead scores.
    ushort_t* swT  = xb;        // written by softmax, after kv GEMM consumed xb
    ushort_t* outb = scoresT;   // written by scatter, after mix GEMM consumed scoresT

    // ---- pack inputs to bf16 -------------------------------------------------
    {
        long n = (long)BN_ * D_;
        pack_n_kernel<<<dim3((unsigned)((n + 255) / 256)), 256, 0, stream>>>(x, xb, n);
    }
    pack_n_kernel<<<dim3(1024), 256, 0, stream>>>(W_mix, Wmix_b, 512L * 512);
    pack_n_kernel<<<dim3(128),  256, 0, stream>>>(wtq, wtq_b, (long)H_ * M_ * HD_);
    pack_t_kernel<<<dim3(2048), 256, 0, stream>>>(W_kv,  Wkv_t,  512, 1024);
    pack_t_kernel<<<dim3(3072), 256, 0, stream>>>(W_qkv, Wqkv_t, 512, 1536);
    pack_t_kernel<<<dim3(1024), 256, 0, stream>>>(W_out, Wout_t, 512, 512);

    // ---- 1) kv = x @ W_kv + b_kv  -> kvb [BN][1024] bf16 --------------------
    gemm_bt_kernel<2, 2, 2, 4, ushort_t, true>
        <<<dim3(BN_ / 64, 1024 / 128, 1), 256, 0, stream>>>(
            xb, 512, Wkv_t, 512, kvb, 1024, b_kv, 512, 1,
            0, 0, 0, 0, 0, 0);

    // ---- 2) scores[b,h][n][m] = xk[n][d] . wtq[h][m][d]  (K=64) -------------
    gemm_bt_kernel<2, 2, 4, 2, ushort_t, false>
        <<<dim3(N_ / 128, 1, B_ * H_), 256, 0, stream>>>(
            kvb, 1024, wtq_b, 64, scoresT, 512, nullptr, 64, H_,
            (long)N_ * 1024, 64, 0, (long)M_ * HD_, (long)N_ * 512, 64);

    // ---- 3) mix: scoresT[n][c] @ W_mix[o][c]^T  (K=512) -> mixT f32 ---------
    gemm_bt_kernel<2, 2, 2, 4, float, false>
        <<<dim3(N_ / 64, 512 / 128, B_), 256, 0, stream>>>(
            scoresT, 512, Wmix_b, 512, mixT, 512, nullptr, 512, 1,
            (long)N_ * 512, 0, 0, 0, (long)N_ * 512, 0);

    // ---- 4) softmax over M within each head -> slice_w bf16 -----------------
    softmax_m_kernel<<<dim3(BN_ / 8), 256, 0, stream>>>(mixT, swT);

    // ---- 5) slice_tok = slice_w @ xv  (K=8192, split) + denominators --------
    slice_tok_kernel<<<dim3(8, 1, 64), 256, 0, stream>>>(swT, kvb, stok_p, sums_p);
    reduce_parts_kernel<<<dim3(1024), 256, 0, stream>>>(stok_p, sums_p, stok, sums);

    // ---- 6) normalize + layernorm -> stok_ln [B*M][512] bf16 ----------------
    norm_ln_kernel<<<dim3(512), 256, 0, stream>>>(stok, sums, ln_g, ln_b, stok_ln);

    // ---- 7) qkv = stok_ln @ W_qkv  (512x512x1536) -> f32 --------------------
    gemm_bt_kernel<2, 2, 2, 4, float, false>
        <<<dim3((B_ * M_) / 64, 1536 / 128, 1), 256, 0, stream>>>(
            stok_ln, 512, Wqkv_t, 512, qkvf, 1536, nullptr, 512, 1,
            0, 0, 0, 0, 0, 0);

    // ---- 8) mini self-attention over 64 tokens ------------------------------
    mini_attn_kernel<<<dim3(64), 256, 0, stream>>>(qkvf, out_tT);

    // ---- 9) scatter: out[n][d] = sum_m slice_w[m][n] * out_tok[m][d] (K=64) -
    gemm_bt_kernel<2, 2, 4, 2, ushort_t, false>
        <<<dim3(N_ / 128, 1, B_ * H_), 256, 0, stream>>>(
            swT, 512, out_tT, 64, outb, 512, nullptr, 64, H_,
            (long)N_ * 512, 64, (long)H_ * 4096, 4096, (long)N_ * 512, 64);

    // ---- 10) final projection: out = outb @ W_out + b_out -> f32 d_out ------
    gemm_bt_kernel<2, 2, 2, 4, float, true>
        <<<dim3(BN_ / 64, 512 / 128, 1), 256, 0, stream>>>(
            outb, 512, Wout_t, 512, out, 512, b_out, 512, 1,
            0, 0, 0, 0, 0, 0);
}
